// PointNet_34059090657432
// MI455X (gfx1250) — compile-verified, matrix operand order checked by reading, not run on hardware
//
#include <hip/hip_runtime.h>
#include <cstdint>
#include <cstddef>

typedef _Float16 h16;
typedef __attribute__((ext_vector_type(16))) _Float16 v16h;
typedef __attribute__((ext_vector_type(8)))  _Float16 v8h;
typedef __attribute__((ext_vector_type(8)))  float    v8f;

#define KNB 64
#define CDIV(a,b) (((a)+(b)-1)/(b))

// ---------------------------------------------------------------------------
// WMMA fragment loader: 16 halves of one row (A) or one weight-row (B column),
// laid out per CDNA5 16-bit A/B layout:
//   lanes 0-15 : elements 0..7 = K(k0..k0+7),   8..15 = K(k0+16..k0+23)
//   lanes 16-31: elements 0..7 = K(k0+8..k0+15),8..15 = K(k0+24..k0+31)
// ---------------------------------------------------------------------------
__device__ __forceinline__ v16h ldfrag(const h16* __restrict__ rowbase, int k, int hi) {
  v8h a = *(const v8h*)(rowbase + k + hi * 8);
  v8h b = *(const v8h*)(rowbase + k + 16 + hi * 8);
  return __builtin_shufflevector(a, b, 0,1,2,3,4,5,6,7,8,9,10,11,12,13,14,15);
}

// ---------------------------------------------------------------------------
// Fused GEMM: out[r,c] = epilogue( sum_k A[r,k]*W[c,k] + bias[c] )
// epilogue: optional ReLU then optional BN: g*(x/sqrt(1+eps)) + beta
// A: [R, lda] f16 row-major (lda multiple of 32), W: [Cout, lda] f16.
// One wave -> 16x64 output tile via 4x v_wmma_f32_16x16x32_f16.
// ---------------------------------------------------------------------------
__global__ __launch_bounds__(256) void wmma_gemm(
    const h16* __restrict__ A, int lda,
    const h16* __restrict__ W,
    const float* __restrict__ bias,
    const float* __restrict__ gamma,
    const float* __restrict__ beta,
    h16* __restrict__ out, int R, int Cout,
    int relu, int bn)
{
  const int lane = threadIdx.x & 31;
  const int wave = threadIdx.x >> 5;
  const int hi   = lane >> 4;
  const int ln   = lane & 15;
  const int row0 = (blockIdx.x * 8 + wave) * 16;
  if (row0 >= R) return;
  const int col0 = blockIdx.y * 64;

  v8f acc[4] = {};
  const v16h zfrag = {};

  const int  arow = row0 + ln;
  const bool av   = (arow < R);
  const h16* Ar = A + (size_t)arow * lda;
  const h16* W0 = W + (size_t)(col0 + ln +  0) * lda;
  const h16* W1 = W + (size_t)(col0 + ln + 16) * lda;
  const h16* W2 = W + (size_t)(col0 + ln + 32) * lda;
  const h16* W3 = W + (size_t)(col0 + ln + 48) * lda;

  for (int k = 0; k < lda; k += 32) {
    v16h af = av ? ldfrag(Ar, k, hi) : zfrag;
    v16h b0 = ldfrag(W0, k, hi);
    v16h b1 = ldfrag(W1, k, hi);
    v16h b2 = ldfrag(W2, k, hi);
    v16h b3 = ldfrag(W3, k, hi);
    if (k + 32 < lda) {
      __builtin_prefetch(Ar + k + 32, 0, 0);
      __builtin_prefetch(W0 + k + 32, 0, 0);
    }
    acc[0] = __builtin_amdgcn_wmma_f32_16x16x32_f16(false, af, false, b0, (short)0, acc[0], false, false);
    acc[1] = __builtin_amdgcn_wmma_f32_16x16x32_f16(false, af, false, b1, (short)0, acc[1], false, false);
    acc[2] = __builtin_amdgcn_wmma_f32_16x16x32_f16(false, af, false, b2, (short)0, acc[2], false, false);
    acc[3] = __builtin_amdgcn_wmma_f32_16x16x32_f16(false, af, false, b3, (short)0, acc[3], false, false);
  }

  const float INVS = 0.9999950000374997f;  // 1/sqrt(1+1e-5)
#pragma unroll
  for (int j = 0; j < 4; ++j) {
    const int c = col0 + j * 16 + ln;
    const float bb = bias[c];
    const float gg = bn ? gamma[c] * INVS : 1.f;
    const float be = bn ? beta[c] : 0.f;
#pragma unroll
    for (int e = 0; e < 8; ++e) {
      const int r = row0 + hi * 8 + e;
      if (r < R) {
        float v = acc[j][e] + bb;
        if (relu) v = fmaxf(v, 0.f);
        v = gg * v + be;
        out[(size_t)r * Cout + c] = (h16)v;
      }
    }
  }
}

// ---------------------------------------------------------------------------
// Weight convert f32 [cout,cin] -> f16 [cout,cinPad] (zero padded K)
// ---------------------------------------------------------------------------
__global__ void convert_w(const float* __restrict__ W, h16* __restrict__ Wh,
                          int cin, int cinPad, int total) {
  int t = blockIdx.x * blockDim.x + threadIdx.x;
  if (t >= total) return;
  int c = t % cinPad, co = t / cinPad;
  Wh[t] = (c < cin) ? (h16)W[(size_t)co * cin + c] : (h16)0.f;
}

// x0 = concat([x, score]) -> f16 [B*N0, 4]
__global__ void build_x0(const float* __restrict__ x, const float* __restrict__ score,
                         h16* __restrict__ x0h, int n) {
  int t = blockIdx.x * blockDim.x + threadIdx.x;
  if (t >= n) return;
  x0h[t*4+0] = (h16)x[t*3+0];
  x0h[t*4+1] = (h16)x[t*3+1];
  x0h[t*4+2] = (h16)x[t*3+2];
  x0h[t*4+3] = (h16)score[t];
}

// ---------------------------------------------------------------------------
// Farthest point sampling, one block per graph. Matches jax scan semantics:
// emit current 'last' (start=0), update dmin with pos[last], argmax -> new last.
// Ties resolve to lowest index (jnp.argmax).
// ---------------------------------------------------------------------------
__global__ __launch_bounds__(256) void fps_kernel(const float* __restrict__ pos, int N, int m,
                                                  int* __restrict__ outIdx, float* __restrict__ qpos) {
  __shared__ float sx[2048], sy[2048], sz[2048], dmin[2048];
  __shared__ float rv[256];
  __shared__ int   ri[256];
  __shared__ int   slast;
  const int b = blockIdx.x, tid = threadIdx.x;
  const float* P = pos + (size_t)b * N * 3;
  for (int j = tid; j < N; j += 256) {
    sx[j] = P[j*3+0]; sy[j] = P[j*3+1]; sz[j] = P[j*3+2];
    dmin[j] = 3.4e38f;
  }
  if (tid == 0) slast = 0;
  __syncthreads();
  int*   oi = outIdx + (size_t)b * m;
  float* oq = qpos   + (size_t)b * m * 3;
  for (int i = 0; i < m; ++i) {
    const int last = slast;
    if (tid == 0) {
      oi[i] = last;
      oq[i*3+0] = sx[last]; oq[i*3+1] = sy[last]; oq[i*3+2] = sz[last];
    }
    const float lx = sx[last], ly = sy[last], lz = sz[last];
    float bv = -1.f; int bi = 0x7fffffff;
    for (int j = tid; j < N; j += 256) {
      float dx = sx[j]-lx, dy = sy[j]-ly, dz = sz[j]-lz;
      float d  = dx*dx + dy*dy + dz*dz;
      float dm = dmin[j];
      if (d < dm) { dm = d; dmin[j] = dm; }
      if (dm > bv) { bv = dm; bi = j; }
    }
    rv[tid] = bv; ri[tid] = bi;
    __syncthreads();
    for (int s = 128; s > 0; s >>= 1) {
      if (tid < s) {
        float v2 = rv[tid+s]; int i2 = ri[tid+s];
        if (v2 > rv[tid] || (v2 == rv[tid] && i2 < ri[tid])) { rv[tid] = v2; ri[tid] = i2; }
      }
      __syncthreads();
    }
    if (tid == 0) slast = ri[0];
    __syncthreads();
  }
}

// ---------------------------------------------------------------------------
// Radius neighbors: thread per query, keep the 64 nearest in-radius points.
// nidx = -1 marks invalid slots.
// ---------------------------------------------------------------------------
__global__ void radius_nn_kernel(const float* __restrict__ pos, const float* __restrict__ qpos,
                                 int* __restrict__ nidx, int B_, int N, int M, float r2) {
  int t = blockIdx.x * blockDim.x + threadIdx.x;
  if (t >= B_ * M) return;
  int b = t / M;
  const float* P = pos + (size_t)b * N * 3;
  float qx = qpos[(size_t)t*3+0], qy = qpos[(size_t)t*3+1], qz = qpos[(size_t)t*3+2];
  float nd[KNB]; int ni[KNB];
  int cnt = 0; float curMax = -1.f; int curPos = 0;
  for (int j = 0; j < N; ++j) {
    float dx = P[j*3+0]-qx, dy = P[j*3+1]-qy, dz = P[j*3+2]-qz;
    float d2 = dx*dx + dy*dy + dz*dz;
    if (d2 > r2) continue;
    if (cnt < KNB) {
      nd[cnt] = d2; ni[cnt] = j;
      if (d2 > curMax) { curMax = d2; curPos = cnt; }
      ++cnt;
    } else if (d2 < curMax) {
      nd[curPos] = d2; ni[curPos] = j;
      curMax = -1.f;
      for (int i = 0; i < KNB; ++i) if (nd[i] > curMax) { curMax = nd[i]; curPos = i; }
    }
  }
  int* o = nidx + (size_t)t * KNB;
  for (int i = 0; i < cnt; ++i) o[i] = ni[i];
  for (int i = cnt; i < KNB; ++i) o[i] = -1;
}

// Gather PointNetConv rows: [feat_j (F) | pos_j - qpos_i (3) | 0 pad] f16
__global__ void gather_sa(const h16* __restrict__ feat, int F,
                          const float* __restrict__ pos, const float* __restrict__ qpos,
                          const int* __restrict__ nidx,
                          h16* __restrict__ A, int lda, int B_, int Nsrc, int M) {
  int r = blockIdx.x * blockDim.x + threadIdx.x;
  int total = B_ * M * KNB;
  if (r >= total) return;
  int b = r / (M * KNB);
  int q = (r / KNB) % M;
  int j = nidx[r];
  h16* Ar = A + (size_t)r * lda;
  if (j < 0) { for (int c = 0; c < lda; ++c) Ar[c] = (h16)0.f; return; }
  const h16* fj = feat + ((size_t)b * Nsrc + j) * F;
  for (int c = 0; c < F; ++c) Ar[c] = fj[c];
  const float* pj = pos + ((size_t)b * Nsrc + j) * 3;
  const float* pq = qpos + ((size_t)b * M + q) * 3;
  Ar[F+0] = (h16)(pj[0]-pq[0]);
  Ar[F+1] = (h16)(pj[1]-pq[1]);
  Ar[F+2] = (h16)(pj[2]-pq[2]);
  for (int c = F + 3; c < lda; ++c) Ar[c] = (h16)0.f;
}

// Masked max over 64 neighbors; empty neighborhood -> 0
__global__ void max_aggregate(const h16* __restrict__ h, const int* __restrict__ nidx,
                              h16* __restrict__ out, int P, int C) {
  int t = blockIdx.x * blockDim.x + threadIdx.x;
  if (t >= P * C) return;
  int p = t / C, c = t % C;
  const int* ni = nidx + (size_t)p * KNB;
  float best = 0.f; bool any = false;
  for (int kn = 0; kn < KNB; ++kn) {
    int j = ni[kn];
    if (j >= 0) {
      float v = (float)h[((size_t)p * KNB + kn) * C + c];
      best = any ? fmaxf(best, v) : v;
      any = true;
    }
  }
  out[t] = (h16)(any ? best : 0.f);
}

// Per-graph column max (sa4 global pooling)
__global__ void col_max(const h16* __restrict__ h, h16* __restrict__ out,
                        int B_, int Mrows, int C) {
  int t = blockIdx.x * blockDim.x + threadIdx.x;
  if (t >= B_ * C) return;
  int b = t / C, c = t % C;
  float best = -3.4e38f;
  for (int i = 0; i < Mrows; ++i)
    best = fmaxf(best, (float)h[((size_t)(b * Mrows + i)) * C + c]);
  out[t] = (h16)best;
}

// sa4 input: [x3 (256) | p3 (3) | 0 pad] -> [rows, 288]
__global__ void build_sa4(const h16* __restrict__ x3h, const float* __restrict__ p3,
                          h16* __restrict__ A, int rows) {
  int t = blockIdx.x * blockDim.x + threadIdx.x;
  if (t >= rows * 288) return;
  int r = t / 288, c = t % 288;
  h16 v;
  if (c < 256)      v = x3h[(size_t)r * 256 + c];
  else if (c < 259) v = (h16)p3[(size_t)r * 3 + (c - 256)];
  else              v = (h16)0.f;
  A[t] = v;
}

// fp3 input: [x4 (1024, broadcast per graph) | x3 (256)] -> [rows, 1280]
__global__ void build_fp3(const h16* __restrict__ x4h, const h16* __restrict__ x3h,
                          h16* __restrict__ A, int rows) {
  int t = blockIdx.x * blockDim.x + threadIdx.x;
  if (t >= rows * 1280) return;
  int r = t / 1280, c = t % 1280;
  int b = r / 164;
  A[t] = (c < 1024) ? x4h[(size_t)b * 1024 + c]
                    : x3h[(size_t)r * 256 + (c - 1024)];
}

// kNN(3) inverse-d^2 interpolation, writes into concat input at colOff
__global__ void knn3_interp(const float* __restrict__ psrc, int Ns,
                            const float* __restrict__ pdst, int Nd,
                            const h16* __restrict__ feat, int C,
                            h16* __restrict__ A, int lda, int colOff, int B_) {
  int t = blockIdx.x * blockDim.x + threadIdx.x;
  if (t >= B_ * Nd) return;
  int b = t / Nd;
  const float* ps = psrc + (size_t)b * Ns * 3;
  float qx = pdst[(size_t)t*3+0], qy = pdst[(size_t)t*3+1], qz = pdst[(size_t)t*3+2];
  float d0 = 3.4e38f, d1 = 3.4e38f, d2 = 3.4e38f;
  int   i0 = 0, i1 = 0, i2 = 0;
  for (int j = 0; j < Ns; ++j) {
    float dx = ps[j*3+0]-qx, dy = ps[j*3+1]-qy, dz = ps[j*3+2]-qz;
    float d = dx*dx + dy*dy + dz*dz;
    if (d < d0)      { d2=d1; i2=i1; d1=d0; i1=i0; d0=d; i0=j; }
    else if (d < d1) { d2=d1; i2=i1; d1=d;  i1=j; }
    else if (d < d2) { d2=d;  i2=j; }
  }
  float w0 = 1.f / fmaxf(d0, 1e-16f);
  float w1 = 1.f / fmaxf(d1, 1e-16f);
  float w2 = 1.f / fmaxf(d2, 1e-16f);
  float ws = w0 + w1 + w2;
  const h16* fa = feat + ((size_t)b * Ns + i0) * C;
  const h16* fb = feat + ((size_t)b * Ns + i1) * C;
  const h16* fc = feat + ((size_t)b * Ns + i2) * C;
  h16* Ar = A + (size_t)t * lda + colOff;
  for (int c = 0; c < C; ++c)
    Ar[c] = (h16)((w0*(float)fa[c] + w1*(float)fb[c] + w2*(float)fc[c]) / ws);
}

// Copy f16 [rows,C] into A at colOff with stride ldd; cols [C,Cpad) zeroed
__global__ void copy_cols(const h16* __restrict__ src, int C,
                          h16* __restrict__ dst, int ldd, int colOff, int Cpad, int rows) {
  int t = blockIdx.x * blockDim.x + threadIdx.x;
  if (t >= rows * Cpad) return;
  int r = t / Cpad, c = t % Cpad;
  dst[(size_t)r * ldd + colOff + c] = (c < C) ? src[(size_t)r * C + c] : (h16)0.f;
}

// Head: lin3 (64->1) + score mask + sigmoid
__global__ void lin3_sigmoid(const h16* __restrict__ h, const float* __restrict__ W3,
                             const float* __restrict__ b3, const float* __restrict__ score,
                             float* __restrict__ out, int n) {
  int t = blockIdx.x * blockDim.x + threadIdx.x;
  if (t >= n) return;
  float acc = b3[0];
  const h16* hr = h + (size_t)t * 64;
  for (int c = 0; c < 64; ++c) acc += (float)hr[c] * W3[c];
  float v = (score[t] > 0.00394f) ? acc : 0.f;
  out[t] = 1.f / (1.f + expf(-v));
}

// ---------------------------------------------------------------------------
extern "C" void kernel_launch(void* const* d_in, const int* in_sizes, int n_in,
                              void* d_out, int out_size, void* d_ws, size_t ws_size,
                              hipStream_t stream) {
  (void)in_sizes; (void)n_in; (void)out_size; (void)ws_size;
  const int B = 4, N0 = 2048, M1 = 1024, M2 = 410, M3 = 164;
  const float* x     = (const float*)d_in[0];
  const float* pos   = (const float*)d_in[1];
  const float* score = (const float*)d_in[2];

  // params tree-flatten order: keys sorted fp0,fp1,fp2,fp3,lin1,lin2,lin3,sa1..sa4;
  // per mlp layer: W, b, beta, g; per lin: W, b.  Leaves start at d_in[3].
  struct LH { int w, cin, cout, cinPad; };
  const LH L[23] = {
    /* 0 sa1 */ {41,   7,  64,  32}, {45,  64,  64,  64}, {49,  64, 128,  64},
    /* 3 sa2 */ {53, 131, 128, 160}, {57, 128, 128, 128}, {61, 128, 256, 128},
    /* 6 sa3 */ {65, 259, 256, 288}, {69, 256, 256, 256}, {73, 256, 256, 256},
    /* 9 sa4 */ {77, 259, 256, 288}, {81, 256, 512, 256}, {85, 512,1024, 512},
    /*12 fp3 */ {27,1280, 256,1280}, {31, 256, 256, 256},
    /*14 fp2 */ {19, 512, 256, 512}, {23, 256, 256, 256},
    /*16 fp1 */ {11, 384, 256, 384}, {15, 256, 128, 256},
    /*18 fp0 */ { 3, 132, 128, 160}, { 7, 128,  64, 128},
    /*20 lin1*/ {35,  64,  64,  64},
    /*21 lin2*/ {37,  64,  64,  64},
    /*22 lin3*/ {39,  64,   1,  64}
  };

  size_t off = 0;
  auto alloc = [&](size_t bytes) -> char* {
    char* p = (char*)d_ws + off;
    off = (off + bytes + 255) & ~(size_t)255;
    return p;
  };

  h16* Wh[22];
  for (int i = 0; i < 22; ++i) Wh[i] = (h16*)alloc((size_t)L[i].cout * L[i].cinPad * 2);
  h16*   x0h = (h16*)alloc((size_t)B*N0*4*2);
  float* p1  = (float*)alloc((size_t)B*M1*3*4);
  float* p2  = (float*)alloc((size_t)B*M2*3*4);
  float* p3  = (float*)alloc((size_t)B*M3*3*4);
  int*   id1 = (int*)alloc((size_t)B*M1*4);
  int*   id2 = (int*)alloc((size_t)B*M2*4);
  int*   id3 = (int*)alloc((size_t)B*M3*4);
  int*   nn1 = (int*)alloc((size_t)B*M1*KNB*4);
  int*   nn2 = (int*)alloc((size_t)B*M2*KNB*4);
  int*   nn3 = (int*)alloc((size_t)B*M3*KNB*4);
  h16*   x1h = (h16*)alloc((size_t)B*M1*128*2);
  h16*   x2h = (h16*)alloc((size_t)B*M2*256*2);
  h16*   x3h = (h16*)alloc((size_t)B*M3*256*2);
  h16*   x4h = (h16*)alloc((size_t)B*1024*2);
  h16*   f3  = (h16*)alloc((size_t)B*M3*256*2);
  h16*   f2  = (h16*)alloc((size_t)B*M2*256*2);
  h16*   f1  = (h16*)alloc((size_t)B*M1*128*2);
  h16*   f0  = (h16*)alloc((size_t)B*N0*64*2);
  h16*   hl2 = (h16*)alloc((size_t)B*N0*64*2);
  h16*   AA  = (h16*)alloc((size_t)36*1024*1024);  // gather / even-layer arena
  h16*   HH  = (h16*)alloc((size_t)68*1024*1024);  // odd-layer arena

  // Convert weights to padded f16
  for (int i = 0; i < 22; ++i) {
    int total = L[i].cout * L[i].cinPad;
    convert_w<<<CDIV(total,256),256,0,stream>>>((const float*)d_in[L[i].w], Wh[i],
                                                L[i].cin, L[i].cinPad, total);
  }

  auto gemm = [&](const h16* Ain, int lda, int li, h16* outp, int R, int relu, int bn) {
    const LH& l = L[li];
    const float* bias = (const float*)d_in[l.w + 1];
    const float* beta = bn ? (const float*)d_in[l.w + 2] : nullptr;
    const float* gam  = bn ? (const float*)d_in[l.w + 3] : nullptr;
    dim3 g((unsigned)CDIV(R,128), (unsigned)(l.cout/64));
    wmma_gemm<<<g,256,0,stream>>>(Ain, lda, Wh[li], bias, gam, beta, outp, R, l.cout, relu, bn);
  };

  build_x0<<<CDIV(B*N0,256),256,0,stream>>>(x, score, x0h, B*N0);

  // FPS hierarchy
  fps_kernel<<<B,256,0,stream>>>(pos, N0, M1, id1, p1);
  fps_kernel<<<B,256,0,stream>>>(p1,  M1, M2, id2, p2);
  fps_kernel<<<B,256,0,stream>>>(p2,  M2, M3, id3, p3);

  // Radius neighborhoods
  radius_nn_kernel<<<CDIV(B*M1,128),128,0,stream>>>(pos, p1, nn1, B, N0, M1,  49.f);
  radius_nn_kernel<<<CDIV(B*M2,128),128,0,stream>>>(p1,  p2, nn2, B, M1, M2, 256.f);
  radius_nn_kernel<<<CDIV(B*M3,128),128,0,stream>>>(p2,  p3, nn3, B, M2, M3, 900.f);

  // ---- sa1: 262144 x (7->64->64->128), masked-max over K=64
  const int R1 = B*M1*KNB;
  gather_sa<<<CDIV(R1,256),256,0,stream>>>(x0h, 4, pos, p1, nn1, AA, 32, B, N0, M1);
  gemm(AA, 32, 0, HH, R1, 1, 1);
  gemm(HH, 64, 1, AA, R1, 1, 1);
  gemm(AA, 64, 2, HH, R1, 1, 1);
  max_aggregate<<<CDIV(B*M1*128,256),256,0,stream>>>(HH, nn1, x1h, B*M1, 128);

  // ---- sa2
  const int R2 = B*M2*KNB;
  gather_sa<<<CDIV(R2,256),256,0,stream>>>(x1h, 128, p1, p2, nn2, AA, 160, B, M1, M2);
  gemm(AA, 160, 3, HH, R2, 1, 1);
  gemm(HH, 128, 4, AA, R2, 1, 1);
  gemm(AA, 128, 5, HH, R2, 1, 1);
  max_aggregate<<<CDIV(B*M2*256,256),256,0,stream>>>(HH, nn2, x2h, B*M2, 256);

  // ---- sa3
  const int R3 = B*M3*KNB;
  gather_sa<<<CDIV(R3,256),256,0,stream>>>(x2h, 256, p2, p3, nn3, AA, 288, B, M2, M3);
  gemm(AA, 288, 6, HH, R3, 1, 1);
  gemm(HH, 256, 7, AA, R3, 1, 1);
  gemm(AA, 256, 8, HH, R3, 1, 1);
  max_aggregate<<<CDIV(B*M3*256,256),256,0,stream>>>(HH, nn3, x3h, B*M3, 256);

  // ---- sa4 global MLP + max pool
  const int R4 = B*M3;  // 656
  build_sa4<<<CDIV(R4*288,256),256,0,stream>>>(x3h, p3, AA, R4);
  gemm(AA, 288,  9, HH, R4, 1, 1);
  gemm(HH, 256, 10, AA, R4, 1, 1);
  gemm(AA, 512, 11, HH, R4, 1, 1);
  col_max<<<CDIV(B*1024,256),256,0,stream>>>(HH, x4h, B, M3, 1024);

  // ---- fp3 (k=1 interp == broadcast of x4)
  build_fp3<<<CDIV(R4*1280,256),256,0,stream>>>(x4h, x3h, AA, R4);
  gemm(AA, 1280, 12, HH, R4, 1, 1);
  gemm(HH,  256, 13, f3, R4, 1, 1);

  // ---- fp2: knn3(p3 -> p2) + concat x2
  knn3_interp<<<CDIV(B*M2,128),128,0,stream>>>(p3, M3, p2, M2, f3, 256, AA, 512, 0, B);
  copy_cols<<<CDIV(B*M2*256,256),256,0,stream>>>(x2h, 256, AA, 512, 256, 256, B*M2);
  gemm(AA, 512, 14, HH, B*M2, 1, 1);
  gemm(HH, 256, 15, f2, B*M2, 1, 1);

  // ---- fp1: knn3(p2 -> p1) + concat x1
  knn3_interp<<<CDIV(B*M1,128),128,0,stream>>>(p2, M2, p1, M1, f2, 256, AA, 384, 0, B);
  copy_cols<<<CDIV(B*M1*128,256),256,0,stream>>>(x1h, 128, AA, 384, 256, 128, B*M1);
  gemm(AA, 384, 16, HH, B*M1, 1, 1);
  gemm(HH, 256, 17, f1, B*M1, 1, 1);

  // ---- fp0: knn3(p1 -> pos) + concat x0 (pad 132->160)
  knn3_interp<<<CDIV(B*N0,128),128,0,stream>>>(p1, M1, pos, N0, f1, 128, AA, 160, 0, B);
  copy_cols<<<CDIV(B*N0*32,256),256,0,stream>>>(x0h, 4, AA, 160, 128, 32, B*N0);
  gemm(AA, 160, 18, HH, B*N0, 1, 1);
  gemm(HH, 128, 19, f0, B*N0, 1, 1);

  // ---- head
  gemm(f0, 64, 20, HH,  B*N0, 1, 0);   // lin1 + ReLU
  gemm(HH, 64, 21, hl2, B*N0, 0, 0);   // lin2
  lin3_sigmoid<<<CDIV(B*N0,256),256,0,stream>>>(hl2, (const float*)d_in[39],
                                                (const float*)d_in[40], score,
                                                (float*)d_out, B*N0);
}
